// Model_13039520711308
// MI455X (gfx1250) — matrix operand order repaired and checked by reading, not compiled, not run on hardware
//
#include <hip/hip_runtime.h>
#include <hip/hip_bf16.h>

// ---------------------------------------------------------------------------
// MI455X (gfx1250) implementation.
//  * All matmul-like work (convs via implicit GEMM, FC, LSTM projections,
//    classifier) runs through v_wmma_f32_16x16x32_f16 (wave32, 16x16 tiles,
//    f32 accumulate), with BN/bias/residual/ReLU fused in the epilogue.
//  * 128x64 block tile, 8 waves, each wave register-blocks a 32x32 output
//    patch -> 4 WMMAs per K-step per wave.
//  * khw==1 fast path (1x1 convs / dense GEMMs): division-free staging with
//    b128 vector loads (K%4==0 alignment-guarded) + global_prefetch of the
//    next K-slab.
//  * Input pointer order: setup_inputs() dict flattened recursively in
//    insertion order (see parsing code below).
// ---------------------------------------------------------------------------

typedef __attribute__((ext_vector_type(16))) _Float16 v16h;
typedef __attribute__((ext_vector_type(8)))  _Float16 v8h;
typedef __attribute__((ext_vector_type(8)))  float    v8f;

union Frag16 { v16h v; v8h h[2]; _Float16 e[16]; };
union H8 { v8h v; _Float16 e[8]; };

#define TILE_M 128
#define TILE_N 64
#define TILE_K 32
#define LDSS   40   // row stride in halves (80B) -> conflict-free ds_load_b128

// ---------------------------------------------------------------------------
// Generic implicit-GEMM conv / dense GEMM with WMMA.
//   C[M,N] = im2col(A)[M,K] x Wt[N,K]^T  (Wt is OIHW flat = [N][K], K=Cin*kh*kw)
//   epilogue: optional BN fold, optional bias, optional residual add
//   (row stride res_stride), optional ReLU.
//   Dense GEMM = Nimg=M, Hin=Win=Hout=Wout=1, Cin=K, kh=kw=1, stride=1, pad=0.
// ---------------------------------------------------------------------------
__global__ __launch_bounds__(256)
void conv_gemm_wmma(const float* __restrict__ A, const float* __restrict__ Wt,
                    float* __restrict__ Cdst,
                    int M, int N, int K,
                    int Nimg, int Hin, int Win, int Cin,
                    int Hout, int Wout, int kh, int kw, int stride, int pad,
                    const float* __restrict__ bng, const float* __restrict__ bnb,
                    const float* __restrict__ bnm, const float* __restrict__ bnv,
                    const float* __restrict__ bias,
                    const float* __restrict__ resid, long long res_stride,
                    int relu)
{
  __shared__ __align__(16) _Float16 As[TILE_M * LDSS];
  __shared__ __align__(16) _Float16 Bs[TILE_N * LDSS];

  const int tid  = threadIdx.x;
  const int lane = tid & 31;
  const int wid  = tid >> 5;   // 8 waves
  const int wm   = wid & 3;    // 4 row bands of 32
  const int wn   = wid >> 2;   // 2 col bands of 32

  const int m0 = blockIdx.y * TILE_M;
  const int n0 = blockIdx.x * TILE_N;

  // A staging: 256 threads, 16 halves each (128 rows x 32 k)
  const int arow = tid >> 1;
  const int acol = (tid & 1) * 16;
  // B staging: 8 halves each (64 rows x 32 k)
  const int brow = tid >> 2;
  const int bcol = (tid & 3) * 8;

  const int am = m0 + arow;
  int img = 0, oy = 0, ox = 0;
  if (am < M) {
    const int hw = Hout * Wout;
    img = am / hw;
    const int rp = am - img * hw;
    oy = rp / Wout;
    ox = rp - oy * Wout;
  }
  const int khw = kh * kw;
  const int oyS = oy * stride - pad;
  const int oxS = ox * stride - pad;
  const bool kvec = ((K & 3) == 0);  // 16B-aligned rows -> b128 staging loads

  v8f acc[2][2] = {};

  for (int k0 = 0; k0 < K; k0 += TILE_K) {
    // ---- stage A tile (f32 -> f16) ----
    {
      _Float16* dst = &As[arow * LDSS + acol];
      if (khw == 1) {
        // 1x1 conv / dense GEMM: contiguous, division-free
        const bool inb = (am < M) && oyS >= 0 && oyS < Hin && oxS >= 0 && oxS < Win;
        const float* src =
            A + (((long long)img * Hin + oyS) * Win + oxS) * Cin + k0 + acol;
        if (inb && (k0 + TILE_K) < K) __builtin_prefetch(src + TILE_K, 0, 2);
        if (inb && kvec && (k0 + acol + 16) <= K) {
          const float4* s4 = (const float4*)src;
          H8 lo, hi;
          #pragma unroll
          for (int j4 = 0; j4 < 2; ++j4) {
            const float4 f = s4[j4];
            lo.e[j4 * 4 + 0] = (_Float16)f.x; lo.e[j4 * 4 + 1] = (_Float16)f.y;
            lo.e[j4 * 4 + 2] = (_Float16)f.z; lo.e[j4 * 4 + 3] = (_Float16)f.w;
          }
          #pragma unroll
          for (int j4 = 0; j4 < 2; ++j4) {
            const float4 f = s4[2 + j4];
            hi.e[j4 * 4 + 0] = (_Float16)f.x; hi.e[j4 * 4 + 1] = (_Float16)f.y;
            hi.e[j4 * 4 + 2] = (_Float16)f.z; hi.e[j4 * 4 + 3] = (_Float16)f.w;
          }
          *(v8h*)(dst)     = lo.v;
          *(v8h*)(dst + 8) = hi.v;
        } else {
          #pragma unroll
          for (int j = 0; j < 16; ++j) {
            const int k = k0 + acol + j;
            dst[j] = (_Float16)((inb && k < K) ? src[j] : 0.f);
          }
        }
      } else {
        int k  = k0 + acol;
        int ci = k / khw;
        int rr = k - ci * khw;
        #pragma unroll
        for (int j = 0; j < 16; ++j) {
          float v = 0.f;
          if (am < M && (k + j) < K) {
            const int ky = rr / kw;
            const int kx = rr - ky * kw;
            const int iy = oyS + ky;
            const int ix = oxS + kx;
            if (iy >= 0 && iy < Hin && ix >= 0 && ix < Win)
              v = A[(((long long)img * Hin + iy) * Win + ix) * Cin + ci];
          }
          dst[j] = (_Float16)v;
          if (++rr == khw) { rr = 0; ++ci; }
        }
      }
    }
    // ---- stage B tile (weights [N][K], f32 -> f16) + L2 prefetch ----
    {
      const int bn = n0 + brow;
      _Float16* dst = &Bs[brow * LDSS + bcol];
      const float* wsrc = Wt + (long long)bn * K + k0 + bcol;
      if (bn < N && (k0 + TILE_K) < K) __builtin_prefetch(wsrc + TILE_K, 0, 2);
      if (bn < N && kvec && (k0 + bcol + 8) <= K) {
        const float4* s4 = (const float4*)wsrc;
        H8 w8;
        #pragma unroll
        for (int j4 = 0; j4 < 2; ++j4) {
          const float4 f = s4[j4];
          w8.e[j4 * 4 + 0] = (_Float16)f.x; w8.e[j4 * 4 + 1] = (_Float16)f.y;
          w8.e[j4 * 4 + 2] = (_Float16)f.z; w8.e[j4 * 4 + 3] = (_Float16)f.w;
        }
        *(v8h*)dst = w8.v;
      } else {
        #pragma unroll
        for (int j = 0; j < 8; ++j) {
          const int k = k0 + bcol + j;
          dst[j] = (_Float16)((bn < N && k < K) ? wsrc[j] : 0.f);
        }
      }
    }
    __syncthreads();

    // ---- A fragments (ISA 7.12.2 16-bit A layout) ----
    Frag16 a0, a1;
    {
      const int kb = (lane >= 16) ? 8 : 0;
      const _Float16* ap0 = &As[(wm * 32 + (lane & 15)) * LDSS];
      a0.h[0] = *(const v8h*)(ap0 + kb);
      a0.h[1] = *(const v8h*)(ap0 + 16 + kb);
      const _Float16* ap1 = &As[(wm * 32 + 16 + (lane & 15)) * LDSS];
      a1.h[0] = *(const v8h*)(ap1 + kb);
      a1.h[1] = *(const v8h*)(ap1 + 16 + kb);
    }
    // ---- B fragments (lane holds col n=lane%16; K = [kb..kb+7|16+kb..16+kb+7], kb=8*(lane/16)) ----
    Frag16 b0, b1;
    {
      const int kb = (lane >= 16) ? 8 : 0;
      const _Float16* bp0 = &Bs[(wn * 32 + (lane & 15)) * LDSS];
      b0.h[0] = *(const v8h*)(bp0 + kb);
      b0.h[1] = *(const v8h*)(bp0 + 16 + kb);
      const _Float16* bp1 = &Bs[(wn * 32 + 16 + (lane & 15)) * LDSS];
      b1.h[0] = *(const v8h*)(bp1 + kb);
      b1.h[1] = *(const v8h*)(bp1 + 16 + kb);
    }
    acc[0][0] = __builtin_amdgcn_wmma_f32_16x16x32_f16(false, a0.v, false, b0.v,
                                                       (short)0, acc[0][0], false, false);
    acc[0][1] = __builtin_amdgcn_wmma_f32_16x16x32_f16(false, a0.v, false, b1.v,
                                                       (short)0, acc[0][1], false, false);
    acc[1][0] = __builtin_amdgcn_wmma_f32_16x16x32_f16(false, a1.v, false, b0.v,
                                                       (short)0, acc[1][0], false, false);
    acc[1][1] = __builtin_amdgcn_wmma_f32_16x16x32_f16(false, a1.v, false, b1.v,
                                                       (short)0, acc[1][1], false, false);
    __syncthreads();
  }

  // ---- epilogue: C layout (ISA 7.12.2): n = lane%16, m = r + (lane>=16?8:0)
  #pragma unroll
  for (int nj = 0; nj < 2; ++nj) {
    const int n = n0 + wn * 32 + nj * 16 + (lane & 15);
    if (n >= N) continue;
    float s = 1.f, off = 0.f;
    if (bng) {
      const float sc = bng[n] * rsqrtf(bnv[n] + 1e-5f);
      s = sc;
      off = bnb[n] - bnm[n] * sc;
    }
    if (bias) off += bias[n];
    #pragma unroll
    for (int mi = 0; mi < 2; ++mi) {
      const int mbase = m0 + wm * 32 + mi * 16 + ((lane >= 16) ? 8 : 0);
      #pragma unroll
      for (int r = 0; r < 8; ++r) {
        const int m = mbase + r;
        if (m >= M) continue;
        float val = acc[mi][nj][r] * s + off;
        if (resid) val += resid[(long long)m * res_stride + n];
        if (relu) val = fmaxf(val, 0.f);
        Cdst[(long long)m * N + n] = val;
      }
    }
  }
}

// ---------------------------------------------------------------------------
// NCHW -> NHWC
// ---------------------------------------------------------------------------
__global__ void nchw_to_nhwc_k(const float* __restrict__ in, float* __restrict__ out,
                               int Nimg, int C, int H, int W)
{
  long long idx = (long long)blockIdx.x * blockDim.x + threadIdx.x;
  const long long total = (long long)Nimg * C * H * W;
  if (idx >= total) return;
  const int w = idx % W;
  const int h = (idx / W) % H;
  const int c = (idx / ((long long)W * H)) % C;
  const int n = idx / ((long long)W * H * C);
  out[(((long long)n * H + h) * W + w) * C + c] = in[idx];
}

// ---------------------------------------------------------------------------
// 3x3 s2 pad1 maxpool, NHWC
// ---------------------------------------------------------------------------
__global__ void maxpool_nhwc_k(const float* __restrict__ in, float* __restrict__ out,
                               int Nimg, int H, int W, int C, int Ho, int Wo)
{
  long long idx = (long long)blockIdx.x * blockDim.x + threadIdx.x;
  const long long total = (long long)Nimg * Ho * Wo * C;
  if (idx >= total) return;
  const int c  = idx % C;
  const int ox = (idx / C) % Wo;
  const int oy = (idx / ((long long)C * Wo)) % Ho;
  const int n  = idx / ((long long)C * Wo * Ho);
  float mx = -INFINITY;
  #pragma unroll
  for (int ky = 0; ky < 3; ++ky) {
    const int iy = oy * 2 - 1 + ky;
    if (iy < 0 || iy >= H) continue;
    #pragma unroll
    for (int kx = 0; kx < 3; ++kx) {
      const int ix = ox * 2 - 1 + kx;
      if (ix < 0 || ix >= W) continue;
      mx = fmaxf(mx, in[(((long long)n * H + iy) * W + ix) * C + c]);
    }
  }
  out[idx] = mx;
}

// ---------------------------------------------------------------------------
// ROI-align (out=3, sampling 2x2), NHWC feature map, writes into the fused
// feature matrix [R, 13824] at column base + c*9 + iy*3 + ix.
// ---------------------------------------------------------------------------
__global__ void roi_align_k(const float* __restrict__ feat, const float* __restrict__ rois,
                            const int* __restrict__ batch_idx, float* __restrict__ out,
                            int H, int W, int C, float scale, int featStride, int base)
{
  const int r = blockIdx.x;
  const int c = blockIdx.y * blockDim.x + threadIdx.x;
  if (c >= C) return;
  const float x1 = rois[r * 4 + 0] * scale;
  const float y1 = rois[r * 4 + 1] * scale;
  const float x2 = rois[r * 4 + 2] * scale;
  const float y2 = rois[r * 4 + 3] * scale;
  const float bw = fmaxf(x2 - x1, 1.f) * (1.f / 3.f);
  const float bh = fmaxf(y2 - y1, 1.f) * (1.f / 3.f);
  const float* fb = feat + (long long)batch_idx[r] * H * W * C;
  for (int iy = 0; iy < 3; ++iy) {
    for (int ix = 0; ix < 3; ++ix) {
      float acc = 0.f;
      #pragma unroll
      for (int sy = 0; sy < 2; ++sy) {
        #pragma unroll
        for (int sx = 0; sx < 2; ++sx) {
          const float gy = ((float)(2 * iy + sy) + 0.5f) * 0.5f;
          const float gx = ((float)(2 * ix + sx) + 0.5f) * 0.5f;
          const float yy = y1 + bh * gy;
          const float xx = x1 + bw * gx;
          const bool valid = (yy >= -1.f) && (yy <= (float)H) &&
                             (xx >= -1.f) && (xx <= (float)W);
          const float y = fminf(fmaxf(yy, 0.f), (float)(H - 1));
          const float x = fminf(fmaxf(xx, 0.f), (float)(W - 1));
          const int y0 = (int)floorf(y), x0 = (int)floorf(x);
          const int y1i = min(y0 + 1, H - 1), x1i = min(x0 + 1, W - 1);
          const float ly = y - (float)y0, lx = x - (float)x0;
          const float v00 = fb[((long long)y0 * W + x0) * C + c];
          const float v01 = fb[((long long)y0 * W + x1i) * C + c];
          const float v10 = fb[((long long)y1i * W + x0) * C + c];
          const float v11 = fb[((long long)y1i * W + x1i) * C + c];
          const float v = v00 * (1.f - ly) * (1.f - lx) + v01 * (1.f - ly) * lx +
                          v10 * ly * (1.f - lx) + v11 * ly * lx;
          acc += valid ? v : 0.f;
        }
      }
      out[(long long)r * featStride + base + c * 9 + iy * 3 + ix] = acc * 0.25f;
    }
  }
}

// ---------------------------------------------------------------------------
// LSTM pointwise cell (mask is all-true for this setup); writes h into lout.
// ---------------------------------------------------------------------------
__device__ __forceinline__ float sigm(float x) { return 1.f / (1.f + __expf(-x)); }

__global__ void lstm_cell_k(const float* __restrict__ gt, float* __restrict__ h,
                            float* __restrict__ c, float* __restrict__ lout,
                            int t, int dirOff)
{
  const int idx = blockIdx.x * blockDim.x + threadIdx.x;
  if (idx >= 32 * 512) return;
  const int s = idx >> 9, j = idx & 511;
  const float* g = gt + (long long)s * 2048;
  const float gi = g[j], gf = g[512 + j], gg = g[1024 + j], go = g[1536 + j];
  const float c2 = sigm(gf) * c[idx] + sigm(gi) * tanhf(gg);
  const float h2 = sigm(go) * tanhf(c2);
  c[idx] = c2;
  h[idx] = h2;
  lout[((long long)s * 16 + t) * 1024 + dirOff + j] = h2;
}

// row scatter/gather via (seq,pos) indices; block per row
__global__ void scatter_rows_k(const float* __restrict__ src, float* __restrict__ dst,
                               const int* __restrict__ seq, const int* __restrict__ pos,
                               int ncols)
{
  const int r = blockIdx.x;
  const long long d = (long long)(seq[r] * 16 + pos[r]);
  for (int c = threadIdx.x; c < ncols; c += blockDim.x)
    dst[d * ncols + c] = src[(long long)r * ncols + c];
}
__global__ void gather_rows_k(const float* __restrict__ src, float* __restrict__ dst,
                              const int* __restrict__ seq, const int* __restrict__ pos,
                              int ncols)
{
  const int r = blockIdx.x;
  const long long s = (long long)(seq[r] * 16 + pos[r]);
  for (int c = threadIdx.x; c < ncols; c += blockDim.x)
    dst[(long long)r * ncols + c] = src[s * ncols + c];
}

__global__ void add_vec_k(const float* __restrict__ a, const float* __restrict__ b,
                          float* __restrict__ o, int n)
{
  const int i = blockIdx.x * blockDim.x + threadIdx.x;
  if (i < n) o[i] = a[i] + b[i];
}

// ===========================================================================
// Host orchestration
// ===========================================================================
struct BNp { const float *g, *b, *m, *v; };
struct Blk { const float* w1; BNp b1; const float* w2; BNp b2;
             const float* w3; BNp b3; const float* wd; BNp bd; };

static void launch_gemm(hipStream_t st, const float* A, const float* Wt, float* C,
                        int M, int N, int K,
                        int Nimg, int Hin, int Win, int Cin,
                        int Hout, int Wout, int kh, int kw, int stride, int pad,
                        const BNp* bn, const float* bias,
                        const float* res, long long res_stride, int relu)
{
  dim3 grid((N + TILE_N - 1) / TILE_N, (M + TILE_M - 1) / TILE_M);
  conv_gemm_wmma<<<grid, 256, 0, st>>>(A, Wt, C, M, N, K,
                                       Nimg, Hin, Win, Cin, Hout, Wout,
                                       kh, kw, stride, pad,
                                       bn ? bn->g : nullptr, bn ? bn->b : nullptr,
                                       bn ? bn->m : nullptr, bn ? bn->v : nullptr,
                                       bias, res, res_stride, relu);
}

static void launch_conv(hipStream_t st, const float* A, const float* Wt, float* C,
                        int Nimg, int Hin, int Win, int Cin, int Cout,
                        int kh, int kw, int stride, int pad,
                        const BNp* bn, const float* res, long long res_stride, int relu)
{
  const int Hout = (Hin + 2 * pad - kh) / stride + 1;
  const int Wout = (Win + 2 * pad - kw) / stride + 1;
  launch_gemm(st, A, Wt, C, Nimg * Hout * Wout, Cout, Cin * kh * kw,
              Nimg, Hin, Win, Cin, Hout, Wout, kh, kw, stride, pad,
              bn, nullptr, res, res_stride, relu);
}

static void launch_fc(hipStream_t st, const float* A, const float* Wt, float* C,
                      int M, int N, int K, const float* bias,
                      const float* res, long long res_stride, int relu)
{
  launch_gemm(st, A, Wt, C, M, N, K, M, 1, 1, K, 1, 1, 1, 1, 1, 0,
              nullptr, bias, res, res_stride, relu);
}

static void bottleneck(hipStream_t st, const Blk& b, const float* x, float* out,
                       float* t1, float* t2, float* sc,
                       int Nimg, int H, int W, int Cin, int Cmid, int Cout, int stride)
{
  const int H2 = (H + 2 - 3) / stride + 1;
  const int W2 = (W + 2 - 3) / stride + 1;
  launch_conv(st, x, b.w1, t1, Nimg, H, W, Cin, Cmid, 1, 1, 1, 0, &b.b1, nullptr, 0, 1);
  launch_conv(st, t1, b.w2, t2, Nimg, H, W, Cmid, Cmid, 3, 3, stride, 1, &b.b2, nullptr, 0, 1);
  const float* res = x;
  if (b.wd) {
    launch_conv(st, x, b.wd, sc, Nimg, H, W, Cin, Cout, 1, 1, stride, 0, &b.bd, nullptr, 0, 0);
    res = sc;
  }
  launch_conv(st, t2, b.w3, out, Nimg, H2, W2, Cmid, Cout, 1, 1, 1, 0, &b.b3, res, Cout, 1);
}

extern "C" void kernel_launch(void* const* d_in, const int* in_sizes, int n_in,
                              void* d_out, int out_size, void* d_ws, size_t ws_size,
                              hipStream_t stream)
{
  (void)in_sizes; (void)n_in; (void)out_size; (void)ws_size;

  // ---- inputs (insertion-order flatten) ----
  const float* x_nchw   = (const float*)d_in[0];   // [2,3,384,384]
  const float* rois     = (const float*)d_in[1];   // [512,4]
  const int*   batchidx = (const int*)d_in[2];
  const int*   seq_ids  = (const int*)d_in[3];
  const int*   pos      = (const int*)d_in[4];

  int p = 5;
  auto F = [&](int i) { return (const float*)d_in[i]; };
  auto rdBN = [&]() { BNp r; r.g = F(p++); r.b = F(p++); r.m = F(p++); r.v = F(p++); return r; };
  auto rdBlk = [&](bool down) {
    Blk b; b.w1 = F(p++); b.b1 = rdBN(); b.w2 = F(p++); b.b2 = rdBN();
    b.w3 = F(p++); b.b3 = rdBN();
    if (down) { b.wd = F(p++); b.bd = rdBN(); } else { b.wd = nullptr; b.bd = BNp{}; }
    return b;
  };
  const float* conv1_w = F(p++);
  BNp bn1 = rdBN();
  Blk L1[3]; for (int i = 0; i < 3; ++i) L1[i] = rdBlk(i == 0);
  Blk L2[4]; for (int i = 0; i < 4; ++i) L2[i] = rdBlk(i == 0);
  Blk L3[6]; for (int i = 0; i < 6; ++i) L3[i] = rdBlk(i == 0);
  const float* fc1_w = F(p++);
  const float* fc1_b = F(p++);
  const float* fco_w = F(p++);
  const float* fco_b = F(p++);
  const float* wih_f = F(p++); const float* whh_f = F(p++);
  const float* bih_f = F(p++); const float* bhh_f = F(p++);
  const float* wih_b = F(p++); const float* whh_b = F(p++);
  const float* bih_b = F(p++); const float* bhh_b = F(p++);

  // ---- workspace layout ----
  float* ws = (float*)d_ws;
  long long cur = 0;
  auto alloc = [&](long long n) { float* r = ws + cur; cur += (n + 63) & ~63LL; return r; };
  const long long BIG = 2LL * 192 * 192 * 64;  // 4,718,592 floats
  float* B0 = alloc(BIG); float* B1 = alloc(BIG); float* B2 = alloc(BIG);
  float* B3 = alloc(BIG); float* B4 = alloc(BIG); float* B5 = alloc(BIG);
  float* FEAT   = alloc(512LL * 13824);
  float* XP     = alloc(512LL * 1024);
  float* HFEAT  = alloc(512LL * 1024);
  float* GXF    = alloc(512LL * 2048);
  float* GXB    = alloc(512LL * 2048);
  float* LOUT   = alloc(512LL * 1024);
  float* FEATS2 = alloc(512LL * 1024);
  float* GT     = alloc(64LL * 2048);
  float* Hbuf   = alloc(32LL * 512);
  float* Cbuf   = alloc(32LL * 512);
  float* BSUM   = alloc(2048);

  // ---- stem: NCHW->NHWC, conv1 7x7 s2 + BN + ReLU, maxpool ----
  {
    const long long n = 2LL * 3 * 384 * 384;
    nchw_to_nhwc_k<<<(n + 255) / 256, 256, 0, stream>>>(x_nchw, B5, 2, 3, 384, 384);
  }
  launch_conv(stream, B5, conv1_w, B0, 2, 384, 384, 3, 64, 7, 7, 2, 3, &bn1, nullptr, 0, 1);
  {
    const long long n = 2LL * 96 * 96 * 64;
    maxpool_nhwc_k<<<(n + 255) / 256, 256, 0, stream>>>(B0, B1, 2, 192, 192, 64, 96, 96);
  }

  // ---- layer1 (96x96, 64 -> 256) ----
  bottleneck(stream, L1[0], B1, B0, B2, B3, B4, 2, 96, 96, 64, 64, 256, 1);
  bottleneck(stream, L1[1], B0, B1, B2, B3, B4, 2, 96, 96, 256, 64, 256, 1);
  bottleneck(stream, L1[2], B1, B0, B2, B3, B4, 2, 96, 96, 256, 64, 256, 1);

  // ---- layer2 (96->48, 256 -> 512); result must survive (xl1 in B0) ----
  bottleneck(stream, L2[0], B0, B1, B2, B3, B4, 2, 96, 96, 256, 128, 512, 2);
  bottleneck(stream, L2[1], B1, B0, B2, B3, B4, 2, 48, 48, 512, 128, 512, 1);
  bottleneck(stream, L2[2], B0, B1, B2, B3, B4, 2, 48, 48, 512, 128, 512, 1);
  bottleneck(stream, L2[3], B1, B0, B2, B3, B4, 2, 48, 48, 512, 128, 512, 1);
  float* XL1 = B0;  // [2,48,48,512]

  // ---- layer3 (48->24, 512 -> 1024), keep away from B0 ----
  bottleneck(stream, L3[0], B0, B1, B2, B3, B4, 2, 48, 48, 512, 256, 1024, 2);
  bottleneck(stream, L3[1], B1, B5, B2, B3, B4, 2, 24, 24, 1024, 256, 1024, 1);
  bottleneck(stream, L3[2], B5, B1, B2, B3, B4, 2, 24, 24, 1024, 256, 1024, 1);
  bottleneck(stream, L3[3], B1, B5, B2, B3, B4, 2, 24, 24, 1024, 256, 1024, 1);
  bottleneck(stream, L3[4], B5, B1, B2, B3, B4, 2, 24, 24, 1024, 256, 1024, 1);
  bottleneck(stream, L3[5], B1, B5, B2, B3, B4, 2, 24, 24, 1024, 256, 1024, 1);
  float* XL2 = B5;  // [2,24,24,1024]

  // ---- ROI align into fused feature matrix [512, 13824] ----
  roi_align_k<<<dim3(512, 2), 256, 0, stream>>>(XL1, rois, batchidx, FEAT,
                                                48, 48, 512, 48.f / 384.f, 13824, 0);
  roi_align_k<<<dim3(512, 4), 256, 0, stream>>>(XL2, rois, batchidx, FEAT,
                                                24, 24, 1024, 24.f / 384.f, 13824, 4608);

  // ---- fc1 + ReLU, scatter to sequence layout ----
  launch_fc(stream, FEAT, fc1_w, HFEAT, 512, 1024, 13824, fc1_b, nullptr, 0, 1);
  hipMemsetAsync(XP, 0, 512LL * 1024 * sizeof(float), stream);
  scatter_rows_k<<<512, 256, 0, stream>>>(HFEAT, XP, seq_ids, pos, 1024);

  // ---- bidirectional LSTM ----
  // forward
  add_vec_k<<<8, 256, 0, stream>>>(bih_f, bhh_f, BSUM, 2048);
  launch_fc(stream, XP, wih_f, GXF, 512, 2048, 1024, BSUM, nullptr, 0, 0);
  hipMemsetAsync(Hbuf, 0, 32LL * 512 * sizeof(float), stream);
  hipMemsetAsync(Cbuf, 0, 32LL * 512 * sizeof(float), stream);
  for (int t = 0; t < 16; ++t) {
    launch_fc(stream, Hbuf, whh_f, GT, 32, 2048, 512, nullptr,
              GXF + (long long)t * 2048, 16LL * 2048, 0);
    lstm_cell_k<<<64, 256, 0, stream>>>(GT, Hbuf, Cbuf, LOUT, t, 0);
  }
  // backward
  add_vec_k<<<8, 256, 0, stream>>>(bih_b, bhh_b, BSUM, 2048);
  launch_fc(stream, XP, wih_b, GXB, 512, 2048, 1024, BSUM, nullptr, 0, 0);
  hipMemsetAsync(Hbuf, 0, 32LL * 512 * sizeof(float), stream);
  hipMemsetAsync(Cbuf, 0, 32LL * 512 * sizeof(float), stream);
  for (int t = 15; t >= 0; --t) {
    launch_fc(stream, Hbuf, whh_b, GT, 32, 2048, 512, nullptr,
              GXB + (long long)t * 2048, 16LL * 2048, 0);
    lstm_cell_k<<<64, 256, 0, stream>>>(GT, Hbuf, Cbuf, LOUT, t, 512);
  }

  // ---- gather feats2, classifier, emit outputs ----
  gather_rows_k<<<512, 256, 0, stream>>>(LOUT, FEATS2, seq_ids, pos, 1024);

  float* out_logits = (float*)d_out;                       // 512 x 4212
  float* out_feats2 = out_logits + 512LL * 4212;           // 512 x 1024
  float* out_rois   = out_feats2 + 512LL * 1024;           // 512 x 4
  launch_fc(stream, FEATS2, fco_w, out_logits, 512, 4212, 1024, fco_b, nullptr, 0, 0);
  hipMemcpyAsync(out_feats2, FEATS2, 512LL * 1024 * sizeof(float),
                 hipMemcpyDeviceToDevice, stream);
  hipMemcpyAsync(out_rois, rois, 512LL * 4 * sizeof(float),
                 hipMemcpyDeviceToDevice, stream);
}
